// SwinBlocks_59133109731668
// MI455X (gfx1250) — matrix operand order repaired and checked by reading, not compiled, not run on hardware
//
#include <hip/hip_runtime.h>
#include <math.h>

// ---------------------------------------------------------------------------
// SwinIR-style RSTB stack for MI455X (gfx1250, wave32, WMMA f16->f32).
// B=2, IMG=256, C=64, WIN=8, HEADS=2, hd=32, NL=5, DEPTH=2, MLP=128.
// Bandwidth-bound -> fuse per-window attention & per-token MLP; all GEMMs via
// v_wmma_f32_16x16x32_f16. Weights pre-swizzled once per launch into
// B-fragment order (f16): one 32B vector load per lane per fragment.
// Wave ids go through readfirstlane so per-wave tile dispatch is scalar.
// ---------------------------------------------------------------------------

typedef __attribute__((ext_vector_type(16))) _Float16 v16h;
typedef __attribute__((ext_vector_type(8)))  _Float16 v8h;
typedef __attribute__((ext_vector_type(8)))  float    v8f;

__device__ __forceinline__ v8f wmma_f16(v16h a, v16h b, v8f c) {
  return __builtin_amdgcn_wmma_f32_16x16x32_f16(false, a, false, b, (short)0, c,
                                                false, false);
}

__device__ __forceinline__ int waveid() {
  return __builtin_amdgcn_readfirstlane((int)(threadIdx.x >> 5));
}

// A fragment (16x32 f16) from row-major f16 LDS. Layout: m = lane&15,
// K = e + 16*hi -> one contiguous 16-half run.
__device__ __forceinline__ v16h frag_a(const _Float16* p, int ld, int m0, int k0) {
  int lane = threadIdx.x & 31;
  int m  = m0 + (lane & 15);
  int hi = (lane & 16) ? 16 : 0;
  const _Float16* q = p + m * ld + k0 + hi;
  v8h lo = *(const v8h*)q;          // K = hi .. hi+7
  v8h hh = *(const v8h*)(q + 8);    // K = hi+8 .. hi+15
  return __builtin_shufflevector(lo, hh, 0, 1, 2, 3, 4, 5, 6, 7,
                                 8, 9, 10, 11, 12, 13, 14, 15);
}

// A fragment reading an 8x8 pixel tile out of a 10x10 halo tile (rconv).
// Tile row m -> halo row ((m>>3)+ky)*10 + (m&7)+kx; ld = 72 halves.
__device__ __forceinline__ v16h frag_a_halo(const _Float16* p, int m0, int k0,
                                            int ky, int kx) {
  int lane = threadIdx.x & 31;
  int m  = m0 + (lane & 15);
  int hr = ((m >> 3) + ky) * 10 + (m & 7) + kx;
  int hi = (lane & 16) ? 16 : 0;
  const _Float16* q = p + hr * 72 + k0 + hi;
  v8h lo = *(const v8h*)q;
  v8h hh = *(const v8h*)(q + 8);
  return __builtin_shufflevector(lo, hh, 0, 1, 2, 3, 4, 5, 6, 7,
                                 8, 9, 10, 11, 12, 13, 14, 15);
}

// B fragment (32x16) with B[k][n] = p[n][k] from f16 LDS (token-major K rows):
// n = lane&15, K = e + 16*(lane>=16) -> one contiguous 16-half run.
__device__ __forceinline__ v16h frag_bT_lds(const _Float16* p, int ld, int k0, int n0) {
  int lane = threadIdx.x & 31;
  int n  = n0 + (lane & 15);
  int hi = (lane & 16) ? 16 : 0;
  const _Float16* q = p + n * ld + k0 + hi;
  v8h lo = *(const v8h*)q;
  v8h hh = *(const v8h*)(q + 8);
  return __builtin_shufflevector(lo, hh, 0, 1, 2, 3, 4, 5, 6, 7,
                                 8, 9, 10, 11, 12, 13, 14, 15);
}

// Pre-swizzled B fragment from global f16: [frag][lane][16] -> one 32B load.
__device__ __forceinline__ v16h frag_b_swz(const _Float16* base, int fragIdx) {
  int lane = threadIdx.x & 31;
  return *((const v16h*)base + (size_t)fragIdx * 32 + lane);
}

__device__ __forceinline__ float wsum32(float v) {
#pragma unroll
  for (int m = 16; m >= 1; m >>= 1) v += __shfl_xor(v, m, 32);
  return v;
}
__device__ __forceinline__ float hmax16(float v) {
#pragma unroll
  for (int m = 8; m >= 1; m >>= 1) v = fmaxf(v, __shfl_xor(v, m, 32));
  return v;
}
__device__ __forceinline__ float hsum16(float v) {
#pragma unroll
  for (int m = 8; m >= 1; m >>= 1) v += __shfl_xor(v, m, 32);
  return v;
}

__device__ __forceinline__ int region9(int gy, int gx) {
  int ry = (gy < 248) ? 0 : ((gy < 252) ? 1 : 2);
  int rx = (gx < 248) ? 0 : ((gx < 252) ? 1 : 2);
  return ry * 3 + rx;
}

// ---------------------------------------------------------------------------
// Weight swizzle: [N,K] f32 (nmat matrices stacked) -> B-fragment-major f16.
// Fragment f = nt*(K/32)+kc; element (lane,e) = W[nt*16+(lane&15)]
//                                               [kc*32+16*(lane>=16)+e].
// ---------------------------------------------------------------------------
__global__ void __launch_bounds__(256) swz_weight_kernel(
    const float* __restrict__ w, _Float16* __restrict__ dst,
    int K, int nfk, int per_mat, int nmat) {
  int idx = blockIdx.x * 256 + threadIdx.x;
  if (idx >= per_mat * nmat) return;
  int mat = idx / per_mat;
  int r = idx - mat * per_mat;
  int e = r & 15, l = (r >> 4) & 31, f = r >> 9;
  int nt = f / nfk, kc = f - nt * nfk;
  int n = nt * 16 + (l & 15);
  int k = kc * 32 + ((l & 16) ? 16 : 0) + e;
  dst[idx] = (_Float16)w[(size_t)mat * per_mat + n * K + k];
}

// rconv weights [5][64][64][3][3] -> per layer, per tap, 8 fragments
// (nt in 0..3, kc in 0..1) with B[ci][co] = w[co][ci][tap].
__global__ void __launch_bounds__(256) swz_conv_kernel(
    const float* __restrict__ w, _Float16* __restrict__ dst) {
  int idx = blockIdx.x * 256 + threadIdx.x;
  if (idx >= 5 * 36864) return;
  int li = idx / 36864;
  int r = idx - li * 36864;
  int tap = r / 4096;
  int r2 = r - tap * 4096;
  int e = r2 & 15, l = (r2 >> 4) & 31, f = r2 >> 9;  // f in 0..7
  int nt = f >> 1, kc = f & 1;
  int co = nt * 16 + (l & 15);
  int ci = kc * 32 + ((l & 16) ? 16 : 0) + e;
  dst[idx] = (_Float16)w[(size_t)li * 36864 + (co * 64 + ci) * 9 + tap];
}

// ---------------------------------------------------------------------------
// Input conv 3->64, writes token-major [B, L, C].
// ---------------------------------------------------------------------------
__global__ void __launch_bounds__(64) conv_in_kernel(
    const float* __restrict__ x, const float* __restrict__ cw,
    const float* __restrict__ cb, float* __restrict__ out) {
  __shared__ float sIn[27];
  int pix = blockIdx.x;
  int bb = pix >> 16;
  int l  = pix & 65535;
  int py = l >> 8, px = l & 255;
  if (threadIdx.x < 27) {
    int ci = threadIdx.x / 9, tap = threadIdx.x % 9;
    int sy = py + tap / 3 - 1, sx = px + tap % 3 - 1;
    float v = 0.f;
    if (sy >= 0 && sy < 256 && sx >= 0 && sx < 256)
      v = x[((size_t)bb * 3 + ci) * 65536 + sy * 256 + sx];
    sIn[threadIdx.x] = v;
  }
  __syncthreads();
  int co = threadIdx.x;
  float acc = cb[co];
#pragma unroll
  for (int j = 0; j < 27; ++j) acc += sIn[j] * cw[co * 27 + j];
  out[((size_t)bb * 65536 + l) * 64 + co] = acc;
}

// ---------------------------------------------------------------------------
// Token LayerNorm over C=64. transpose_out=1 writes NCHW (final norm).
// ---------------------------------------------------------------------------
__global__ void __launch_bounds__(256) ln_token_kernel(
    const float* __restrict__ tin, float* __restrict__ tout,
    const float* __restrict__ g, const float* __restrict__ bta,
    int transpose_out) {
  int lane = threadIdx.x & 31, wid = waveid();
  int tok0 = blockIdx.x * 64;
  for (int tk = wid * 8; tk < wid * 8 + 8; ++tk) {
    size_t gt = (size_t)(tok0 + tk);
    const float* row = tin + gt * 64;
    float v0 = row[lane], v1 = row[lane + 32];
    float mean = wsum32(v0 + v1) * (1.f / 64.f);
    float d0 = v0 - mean, d1 = v1 - mean;
    float var = wsum32(d0 * d0 + d1 * d1) * (1.f / 64.f);
    float rs = rsqrtf(var + 1e-5f);
    float y0 = d0 * rs * g[lane] + bta[lane];
    float y1 = d1 * rs * g[lane + 32] + bta[lane + 32];
    if (!transpose_out) {
      tout[gt * 64 + lane] = y0;
      tout[gt * 64 + lane + 32] = y1;
    } else {
      int bb = (int)(gt >> 16);
      int l  = (int)(gt & 65535);
      tout[((size_t)bb * 64 + lane) * 65536 + l] = y0;
      tout[((size_t)bb * 64 + lane + 32) * 65536 + l] = y1;
    }
  }
}

// ---------------------------------------------------------------------------
// Fused windowed attention. One workgroup (8 waves) per 8x8 window.
// ---------------------------------------------------------------------------
__global__ void __launch_bounds__(256) swin_attn_kernel(
    const float* __restrict__ tin, float* __restrict__ tout,
    const float* __restrict__ n1g, const float* __restrict__ n1b,
    const _Float16* __restrict__ qwh, const float* __restrict__ qb,
    const float* __restrict__ rpb, const _Float16* __restrict__ pwh,
    const float* __restrict__ pb, int shift) {
  __shared__ __align__(16) _Float16 sX[64][72];    // LN'd tokens
  __shared__ __align__(16) _Float16 sQK[64][136];  // q(scaled)|k, cols [0,128)
  __shared__ __align__(16) _Float16 sVt[64][72];   // V transposed [dim][token]
  __shared__ __align__(16) _Float16 sP[2][64][72]; // softmax probs per head
  __shared__ __align__(16) _Float16 sO[64][72];    // attn output (heads concat)
  __shared__ float sRPB[450];                      // rel-pos bias [225][2]

  int lane = threadIdx.x & 31, wid = waveid();
  int bb = blockIdx.x >> 10;
  int w  = blockIdx.x & 1023;
  int wr = w >> 5, wc = w & 31;

  // ---- Phase 1: stage rpb + LayerNorm into f16 LDS ----
  for (int i = threadIdx.x; i < 450; i += 256) sRPB[i] = rpb[i];
  for (int tk = wid * 8; tk < wid * 8 + 8; ++tk) {
    int mr = tk >> 3, mc = tk & 7;
    int gy = (wr * 8 + mr + shift) & 255;
    int gx = (wc * 8 + mc + shift) & 255;
    const float* row = tin + (((size_t)bb << 16) + (size_t)(gy * 256 + gx)) * 64;
    float v0 = row[lane], v1 = row[lane + 32];
    float mean = wsum32(v0 + v1) * (1.f / 64.f);
    float d0 = v0 - mean, d1 = v1 - mean;
    float var = wsum32(d0 * d0 + d1 * d1) * (1.f / 64.f);
    float rs = rsqrtf(var + 1e-5f);
    sX[tk][lane]      = (_Float16)(d0 * rs * n1g[lane] + n1b[lane]);
    sX[tk][lane + 32] = (_Float16)(d1 * rs * n1g[lane + 32] + n1b[lane + 32]);
  }
  __syncthreads();

  // ---- Phase 2: QKV = X @ Wqkv^T (48 tiles, 6 per wave) ----
  for (int t = wid; t < 48; t += 8) {
    int mt = t & 3, nt = t >> 2;
    v8f c = {};
#pragma unroll
    for (int ks = 0; ks < 64; ks += 32) {
      v16h a = frag_a(&sX[0][0], 72, mt * 16, ks);
      v16h b = frag_b_swz(qwh, nt * 2 + (ks >> 5));
      c = wmma_f16(a, b, c);
    }
    int n16 = lane & 15, hi = lane >> 4;
    int col = nt * 16 + n16;
    float bias = qb[col];
    if (nt < 4) {        // q, fold hd^-0.5
#pragma unroll
      for (int r = 0; r < 8; ++r)
        sQK[mt * 16 + r + hi * 8][col] =
            (_Float16)((c[r] + bias) * 0.17677669529663687f);
    } else if (nt < 8) { // k
#pragma unroll
      for (int r = 0; r < 8; ++r)
        sQK[mt * 16 + r + hi * 8][col] = (_Float16)(c[r] + bias);
    } else {             // v, store transposed [dim][token]
#pragma unroll
      for (int r = 0; r < 8; ++r)
        sVt[col - 128][mt * 16 + r + hi * 8] = (_Float16)(c[r] + bias);
    }
  }
  __syncthreads();

  // ---- Phase 3: scores + softmax in registers. Wave = (head, 16-row strip). ----
  {
    int head = wid >> 2, mt = wid & 3, m0 = mt * 16;
    v16h aq = frag_a(&sQK[0][head * 32], 136, m0, 0);  // K = hd = 32
    v8f sc[4];
#pragma unroll
    for (int nt = 0; nt < 4; ++nt) {
      v16h bk = frag_bT_lds(&sQK[0][64 + head * 32], 136, 0, nt * 16);
      v8f z = {};
      sc[nt] = wmma_f16(aq, bk, z);
    }
    int n16 = lane & 15, hi = lane >> 4;
#pragma unroll
    for (int nt = 0; nt < 4; ++nt) {
#pragma unroll
      for (int r = 0; r < 8; ++r) {
        int m = m0 + r + hi * 8;
        int n = nt * 16 + n16;
        int mr = m >> 3, mc = m & 7, nr = n >> 3, nc = n & 7;
        int ridx = (mr - nr + 7) * 15 + (mc - nc + 7);
        float v = sc[nt][r] + sRPB[ridx * 2 + head];
        if (shift) {
          if (region9(wr * 8 + mr, wc * 8 + mc) !=
              region9(wr * 8 + nr, wc * 8 + nc))
            v -= 100.f;
        }
        sc[nt][r] = v;
      }
    }
#pragma unroll
    for (int r = 0; r < 8; ++r) {
      float mx = fmaxf(fmaxf(sc[0][r], sc[1][r]), fmaxf(sc[2][r], sc[3][r]));
      mx = hmax16(mx);
      float sum = 0.f;
#pragma unroll
      for (int nt = 0; nt < 4; ++nt) {
        float e = __expf(sc[nt][r] - mx);
        sc[nt][r] = e;
        sum += e;
      }
      sum = hsum16(sum);
      float inv = 1.f / sum;
      int m = m0 + r + hi * 8;
#pragma unroll
      for (int nt = 0; nt < 4; ++nt)
        sP[head][m][nt * 16 + n16] = (_Float16)(sc[nt][r] * inv);
    }
  }
  __syncthreads();

  // ---- Phase 4: O = P @ V (16 tiles, 2 per wave) ----
#pragma unroll
  for (int i = 0; i < 2; ++i) {
    int t = wid * 2 + i;
    int head = t >> 3, rem = t & 7, mt = rem >> 1, nt = rem & 1;
    v8f c = {};
#pragma unroll
    for (int ks = 0; ks < 64; ks += 32) {
      v16h a = frag_a(&sP[head][0][0], 72, mt * 16, ks);
      v16h b = frag_bT_lds(&sVt[head * 32][0], 72, ks, nt * 16);
      c = wmma_f16(a, b, c);
    }
    int n16 = lane & 15, hi = lane >> 4;
#pragma unroll
    for (int r = 0; r < 8; ++r)
      sO[mt * 16 + r + hi * 8][head * 32 + nt * 16 + n16] = (_Float16)c[r];
  }
  __syncthreads();

  // ---- Phase 5: proj + residual + store (16 tiles, 2 per wave) ----
#pragma unroll
  for (int i = 0; i < 2; ++i) {
    int t = wid * 2 + i;
    int mt = t >> 2, nt = t & 3;
    v8f c = {};
#pragma unroll
    for (int ks = 0; ks < 64; ks += 32) {
      v16h a = frag_a(&sO[0][0], 72, mt * 16, ks);
      v16h b = frag_b_swz(pwh, nt * 2 + (ks >> 5));
      c = wmma_f16(a, b, c);
    }
    int n16 = lane & 15, hi = lane >> 4;
    int col = nt * 16 + n16;
    float pbias = pb[col];
#pragma unroll
    for (int r = 0; r < 8; ++r) {
      int m = mt * 16 + r + hi * 8;
      int mr = m >> 3, mc = m & 7;
      int gy = (wr * 8 + mr + shift) & 255;
      int gx = (wc * 8 + mc + shift) & 255;
      size_t off = (((size_t)bb << 16) + (size_t)(gy * 256 + gx)) * 64 + col;
      tout[off] = tin[off] + c[r] + pbias;
    }
  }
}

// ---------------------------------------------------------------------------
// Fused MLP: LN2 + fc1 + exact GeLU + fc2 + residual, in place, 64 tokens/block.
// ---------------------------------------------------------------------------
__global__ void __launch_bounds__(256) swin_mlp_kernel(
    float* __restrict__ t,
    const float* __restrict__ n2g, const float* __restrict__ n2b,
    const _Float16* __restrict__ f1wh, const float* __restrict__ f1b,
    const _Float16* __restrict__ f2wh, const float* __restrict__ f2b) {
  __shared__ __align__(16) _Float16 sX[64][72];
  __shared__ __align__(16) _Float16 sH[64][136];
  int lane = threadIdx.x & 31, wid = waveid();
  size_t tok0 = (size_t)blockIdx.x * 64;

  for (int tk = wid * 8; tk < wid * 8 + 8; ++tk) {
    const float* row = t + (tok0 + tk) * 64;
    float v0 = row[lane], v1 = row[lane + 32];
    float mean = wsum32(v0 + v1) * (1.f / 64.f);
    float d0 = v0 - mean, d1 = v1 - mean;
    float var = wsum32(d0 * d0 + d1 * d1) * (1.f / 64.f);
    float rs = rsqrtf(var + 1e-5f);
    sX[tk][lane]      = (_Float16)(d0 * rs * n2g[lane] + n2b[lane]);
    sX[tk][lane + 32] = (_Float16)(d1 * rs * n2g[lane + 32] + n2b[lane + 32]);
  }
  __syncthreads();

  // fc1 + GeLU: 32 tiles (4 per wave)
  for (int tt = wid; tt < 32; tt += 8) {
    int mt = tt & 3, nt = tt >> 2;
    v8f c = {};
#pragma unroll
    for (int ks = 0; ks < 64; ks += 32) {
      v16h a = frag_a(&sX[0][0], 72, mt * 16, ks);
      v16h b = frag_b_swz(f1wh, nt * 2 + (ks >> 5));
      c = wmma_f16(a, b, c);
    }
    int n16 = lane & 15, hi = lane >> 4;
    int col = nt * 16 + n16;
    float b1 = f1b[col];
#pragma unroll
    for (int r = 0; r < 8; ++r) {
      float v = c[r] + b1;
      float gl = 0.5f * v * (1.f + erff(v * 0.70710678118654752f));
      sH[mt * 16 + r + hi * 8][col] = (_Float16)gl;
    }
  }
  __syncthreads();

  // fc2 + residual: 16 tiles (2 per wave), K=128
#pragma unroll
  for (int i = 0; i < 2; ++i) {
    int tt = wid * 2 + i;
    int mt = tt >> 2, nt = tt & 3;
    v8f c = {};
#pragma unroll
    for (int ks = 0; ks < 128; ks += 32) {
      v16h a = frag_a(&sH[0][0], 136, mt * 16, ks);
      v16h b = frag_b_swz(f2wh, nt * 4 + (ks >> 5));
      c = wmma_f16(a, b, c);
    }
    int n16 = lane & 15, hi = lane >> 4;
    int col = nt * 16 + n16;
    float b2 = f2b[col];
#pragma unroll
    for (int r = 0; r < 8; ++r) {
      int m = mt * 16 + r + hi * 8;
      size_t off = (tok0 + m) * 64 + col;
      t[off] = t[off] + c[r] + b2;
    }
  }
}

// ---------------------------------------------------------------------------
// Residual 3x3 conv (64->64): stage a 10x10 halo tile once, then 9 shifted
// WMMA GEMMs reading the halo; + layer residual. One barrier total.
// ---------------------------------------------------------------------------
__global__ void __launch_bounds__(256) rconv_kernel(
    const float* __restrict__ tin, float* __restrict__ tres,
    const _Float16* __restrict__ wh, const float* __restrict__ bias) {
  __shared__ __align__(16) _Float16 sHalo[100][72];  // 10x10 pixels x 64 ch
  int lane = threadIdx.x & 31, wid = waveid();
  int bb = blockIdx.x >> 10;
  int rem = blockIdx.x & 1023;
  int ty = rem >> 5, tx = rem & 31;

  // stage halo (zero-padded at image borders)
  for (int i = threadIdx.x; i < 6400; i += 256) {
    int p = i >> 6, cc = i & 63;
    int hy = p / 10, hx = p - hy * 10;
    int py = ty * 8 + hy - 1, px = tx * 8 + hx - 1;
    float v = 0.f;
    if (py >= 0 && py < 256 && px >= 0 && px < 256)
      v = tin[(((size_t)bb << 16) + (size_t)(py * 256 + px)) * 64 + cc];
    sHalo[p][cc] = (_Float16)v;
  }
  __syncthreads();

  v8f acc[2];
  acc[0] = {};
  acc[1] = {};
  int mt0 = (wid * 2) >> 2, nt0 = (wid * 2) & 3;
  int mt1 = (wid * 2 + 1) >> 2, nt1 = (wid * 2 + 1) & 3;

#pragma unroll
  for (int tap = 0; tap < 9; ++tap) {
    int ky = tap / 3, kx = tap % 3;  // halo offset already includes -1
#pragma unroll
    for (int ks = 0; ks < 64; ks += 32) {
      v16h a0 = frag_a_halo(&sHalo[0][0], mt0 * 16, ks, ky, kx);
      v16h b0 = frag_b_swz(wh, tap * 8 + nt0 * 2 + (ks >> 5));
      acc[0] = wmma_f16(a0, b0, acc[0]);
      v16h a1 = frag_a_halo(&sHalo[0][0], mt1 * 16, ks, ky, kx);
      v16h b1 = frag_b_swz(wh, tap * 8 + nt1 * 2 + (ks >> 5));
      acc[1] = wmma_f16(a1, b1, acc[1]);
    }
  }

#pragma unroll
  for (int i = 0; i < 2; ++i) {
    int tt = wid * 2 + i;
    int mt = tt >> 2, nt = tt & 3;
    int n16 = lane & 15, hi = lane >> 4;
    int co = nt * 16 + n16;
    float cb = bias[co];
#pragma unroll
    for (int r = 0; r < 8; ++r) {
      int p = mt * 16 + r + hi * 8;
      int l = (ty * 8 + (p >> 3)) * 256 + tx * 8 + (p & 7);
      size_t off = (((size_t)bb << 16) + (size_t)l) * 64 + co;
      tres[off] = tres[off] + acc[i][r] + cb;
    }
  }
}

// ---------------------------------------------------------------------------
extern "C" void kernel_launch(void* const* d_in, const int* in_sizes, int n_in,
                              void* d_out, int out_size, void* d_ws, size_t ws_size,
                              hipStream_t stream) {
  (void)in_sizes; (void)n_in; (void)out_size; (void)ws_size;

  const float* x      = (const float*)d_in[0];
  const float* conv_w = (const float*)d_in[1];
  const float* conv_b = (const float*)d_in[2];
  const float* pe_g   = (const float*)d_in[3];
  const float* pe_b   = (const float*)d_in[4];
  const float* n1_g   = (const float*)d_in[5];
  const float* n1_b   = (const float*)d_in[6];
  const float* qkv_w  = (const float*)d_in[7];
  const float* qkv_b  = (const float*)d_in[8];
  const float* rpb    = (const float*)d_in[9];
  const float* proj_w = (const float*)d_in[10];
  const float* proj_b = (const float*)d_in[11];
  const float* n2_g   = (const float*)d_in[12];
  const float* n2_b   = (const float*)d_in[13];
  const float* fc1_w  = (const float*)d_in[14];
  const float* fc1_b  = (const float*)d_in[15];
  const float* fc2_w  = (const float*)d_in[16];
  const float* fc2_b  = (const float*)d_in[17];
  const float* rc_w   = (const float*)d_in[18];
  const float* rc_b   = (const float*)d_in[19];
  const float* norm_g = (const float*)d_in[20];
  const float* norm_b = (const float*)d_in[21];

  const size_t NTOK = (size_t)2 * 65536;  // B*L
  float* buf0 = (float*)d_ws;             // "cur" (layer residual)
  float* buf1 = buf0 + NTOK * 64;         // "nxt"
  _Float16* wq_h = (_Float16*)(buf1 + NTOK * 64);  // 10 * 192*64
  _Float16* wp_h = wq_h + 10 * 12288;              // 10 * 64*64
  _Float16* w1_h = wp_h + 10 * 4096;               // 10 * 128*64
  _Float16* w2_h = w1_h + 10 * 8192;               // 10 * 64*128
  _Float16* wc_h = w2_h + 10 * 8192;               // 5 * 64*64*9

  // One-shot weight swizzles (f32 -> f16, B-fragment order).
  swz_weight_kernel<<<(10 * 12288 + 255) / 256, 256, 0, stream>>>(
      qkv_w, wq_h, 64, 2, 12288, 10);
  swz_weight_kernel<<<(10 * 4096 + 255) / 256, 256, 0, stream>>>(
      proj_w, wp_h, 64, 2, 4096, 10);
  swz_weight_kernel<<<(10 * 8192 + 255) / 256, 256, 0, stream>>>(
      fc1_w, w1_h, 64, 2, 8192, 10);
  swz_weight_kernel<<<(10 * 8192 + 255) / 256, 256, 0, stream>>>(
      fc2_w, w2_h, 128, 4, 8192, 10);
  swz_conv_kernel<<<(5 * 36864 + 255) / 256, 256, 0, stream>>>(rc_w, wc_h);

  // conv(3->64) -> token-major buf1, then patch-embed LN -> buf0
  conv_in_kernel<<<2 * 65536, 64, 0, stream>>>(x, conv_w, conv_b, buf1);
  ln_token_kernel<<<2048, 256, 0, stream>>>(buf1, buf0, pe_g, pe_b, 0);

  for (int li = 0; li < 5; ++li) {
    for (int d = 0; d < 2; ++d) {
      int bi = li * 2 + d;
      int shift = d ? 4 : 0;
      const float* tin = (d == 0) ? buf0 : buf1;  // block0 preserves buf0 residual
      swin_attn_kernel<<<2048, 256, 0, stream>>>(
          tin, buf1,
          n1_g + bi * 64, n1_b + bi * 64,
          wq_h + (size_t)bi * 12288, qkv_b + bi * 192,
          rpb + (size_t)bi * 450,
          wp_h + (size_t)bi * 4096, proj_b + bi * 64, shift);
      swin_mlp_kernel<<<2048, 256, 0, stream>>>(
          buf1,
          n2_g + bi * 64, n2_b + bi * 64,
          w1_h + (size_t)bi * 8192, fc1_b + bi * 128,
          w2_h + (size_t)bi * 8192, fc2_b + bi * 64);
    }
    rconv_kernel<<<2048, 256, 0, stream>>>(
        buf1, buf0, wc_h + (size_t)li * 36864, rc_b + li * 64);
  }

  // final LN + transpose to NCHW
  ln_token_kernel<<<2048, 256, 0, stream>>>(buf0, (float*)d_out, norm_g, norm_b, 1);
}
